// ViSNetBlock_12695923327641
// MI455X (gfx1250) — matrix operand order repaired and checked by reading, not compiled, not run on hardware
//
#include <hip/hip_runtime.h>
#include <hip/hip_bf16.h>
#include <math.h>

#define N_NODES 16384
#define N_EDGES 131072
#define HDIM 128

typedef __bf16 bfloat;
typedef bfloat v16bf __attribute__((ext_vector_type(16)));
typedef bfloat bf8v  __attribute__((ext_vector_type(8)));
typedef bfloat bf4v  __attribute__((ext_vector_type(4)));
typedef float  v8f   __attribute__((ext_vector_type(8)));

// Fast silu: x * sigmoid(x) with single-instruction v_rcp_f32 (no IEEE div chain).
__device__ __forceinline__ float silu_f(float x) {
  return x * __builtin_amdgcn_rcpf(1.0f + __expf(-x));
}

// ---------------------------------------------------------------------------
// One 16x16 f32 output tile via v_wmma_f32_16x16x32_bf16, K unrolled.
// A: row-major [rows x lda], rows are M.  B: row-major weights W[j][k] so that
// out[i][j] = sum_k A[i][k] * W[j][k]  (i.e. A @ W^T).
// Per ISA 7.12.2 layouts:
//   A lane m(+16hi): elems 0-7 = k0+8hi+0..7, elems 8-15 = k0+8hi+16..23
//   B lane n(+16hi): elems 0-7 = k0+8hi+0..7, elems 8-15 = k0+8hi+16..23 of column n  (= W[n][k] row)
// Both are contiguous in k -> two 16B loads each.
// ---------------------------------------------------------------------------
template <int K>
__device__ __forceinline__ v8f wmma_bf16_tile(const bfloat* __restrict__ A, int lda, int row0,
                                              const bfloat* __restrict__ B, int ldb, int col0,
                                              v8f acc) {
  const int lane = threadIdx.x & 31;
  const int m = lane & 15;
  const int hi = lane >> 4;
  const bfloat* arow = A + (size_t)(row0 + m) * lda;
  const bfloat* brow = B + (size_t)(col0 + m) * ldb;
#pragma unroll
  for (int k0 = 0; k0 < K; k0 += 32) {
    union { v16bf v; bf8v h[2]; } au, bu;
    au.h[0] = *(const bf8v*)(arow + k0 + 8 * hi);
    au.h[1] = *(const bf8v*)(arow + k0 + 8 * hi + 16);
    bu.h[0] = *(const bf8v*)(brow + k0 + 8 * hi);
    bu.h[1] = *(const bf8v*)(brow + k0 + 8 * hi + 16);
    acc = __builtin_amdgcn_wmma_f32_16x16x32_bf16(false, au.v, false, bu.v,
                                                  (short)0, acc, false, false);
  }
  return acc;
}

// ---------------------------------------------------------------------------
__global__ void k_cvt(const float* __restrict__ s, bfloat* __restrict__ d, int n) {
  int i = blockIdx.x * 256 + threadIdx.x;
  if (i < n) d[i] = (bfloat)s[i];
}

// LayerNorm over H=128, one row per wave (8 rows / 256-thread block), bf16 out.
__global__ void k_ln(const float* __restrict__ x, const float* __restrict__ g,
                     const float* __restrict__ b, bfloat* __restrict__ xn) {
  const int row = blockIdx.x * 8 + (threadIdx.x >> 5);
  const int lane = threadIdx.x & 31;
  const float4 v = *(const float4*)(x + (size_t)row * HDIM + lane * 4);
  float s = v.x + v.y + v.z + v.w;
#pragma unroll
  for (int o = 16; o >= 1; o >>= 1) s += __shfl_xor(s, o, 32);
  const float mu = s * (1.0f / 128.0f);
  const float e0 = v.x - mu, e1 = v.y - mu, e2 = v.z - mu, e3 = v.w - mu;
  float q = e0 * e0 + e1 * e1 + e2 * e2 + e3 * e3;
#pragma unroll
  for (int o = 16; o >= 1; o >>= 1) q += __shfl_xor(q, o, 32);
  const float rstd = __builtin_amdgcn_rsqf(q * (1.0f / 128.0f) + 1e-5f);
  const int h = lane * 4;
  const float4 gg = *(const float4*)(g + h);
  const float4 bb = *(const float4*)(b + h);
  bf4v o4;
  o4[0] = (bfloat)(e0 * rstd * gg.x + bb.x);
  o4[1] = (bfloat)(e1 * rstd * gg.y + bb.y);
  o4[2] = (bfloat)(e2 * rstd * gg.z + bb.z);
  o4[3] = (bfloat)(e3 * rstd * gg.w + bb.w);
  *(bf4v*)(xn + (size_t)row * HDIM + h) = o4;
}

// GEMM rows x 384: out = act(A @ W^T + bias); 3 bias segments of 128 cols.
__global__ void k_gemm384(const bfloat* __restrict__ A, const bfloat* __restrict__ W,
                          const float* __restrict__ b0, const float* __restrict__ b1,
                          const float* __restrict__ b2, bfloat* __restrict__ Out,
                          int do_silu) {
  const int row0 = blockIdx.x * 16;
  const int ct = blockIdx.y * 4 + (threadIdx.x >> 5);   // 0..23
  const int lane = threadIdx.x & 31;
  const int n = lane & 15, hi = lane >> 4;
  v8f acc = {};
  acc = wmma_bf16_tile<HDIM>(A, HDIM, row0, W, HDIM, ct * 16, acc);
  const int col = ct * 16 + n;
  const float* bp = (col < 128) ? b0 : (col < 256) ? b1 : b2;
  const float bias = bp[col & 127];
#pragma unroll
  for (int i = 0; i < 8; ++i) {
    float vv = acc[i] + bias;
    if (do_silu) vv = silu_f(vv);
    Out[(size_t)(row0 + i + 8 * hi) * 384 + col] = (bfloat)vv;
  }
}

// vec(bf16)[N,3,128] @ [Wvec;Wwtrg;Wwsrc]^T (640 cols). Fuses vec_dot, writes
// vec3 (f32) and wt/ws (bf16). 8 waves: pair tiles (w, w+8) + 3 tail tiles.
__global__ void k_node_vec(const bfloat* __restrict__ vecb, const bfloat* __restrict__ Wvw,
                           float* __restrict__ vecdot, float* __restrict__ vec3,
                           bfloat* __restrict__ wt, bfloat* __restrict__ wsb) {
  const int row0 = blockIdx.x * 16;
  const int w = threadIdx.x >> 5;
  const int lane = threadIdx.x & 31;
  const int n = lane & 15, hi = lane >> 4;
  v8f vd = {};
  for (int c = 0; c < 3; ++c) {
    const bfloat* A = vecb + c * HDIM;  // row pitch 384: row = node, + c*128
    v8f zero = {};
    v8f t1 = wmma_bf16_tile<HDIM>(A, 384, row0, Wvw, HDIM, w * 16, zero);
    v8f t2 = wmma_bf16_tile<HDIM>(A, 384, row0, Wvw, HDIM, 128 + w * 16, zero);
#pragma unroll
    for (int i = 0; i < 8; ++i) vd[i] += t1[i] * t2[i];
#pragma unroll
    for (int t = 0; t < 3; ++t) {
      const int col0 = (16 + w * 3 + t) * 16;  // tiles 16..39 -> cols 256..639
      v8f o = wmma_bf16_tile<HDIM>(A, 384, row0, Wvw, HDIM, col0, zero);
      const int col = col0 + n;
#pragma unroll
      for (int i = 0; i < 8; ++i) {
        const size_t r = (size_t)(row0 + i + 8 * hi) * 384 + c * HDIM;
        if (col < 384)      vec3[r + (col - 256)] = o[i];
        else if (col < 512) wt[r + (col - 384)] = (bfloat)o[i];
        else                wsb[r + (col - 512)] = (bfloat)o[i];
      }
    }
  }
#pragma unroll
  for (int i = 0; i < 8; ++i)
    vecdot[(size_t)(row0 + i + 8 * hi) * HDIM + w * 16 + n] = vd[i];
}

// Per-edge attention + message: one edge per wave (4 h per lane).
__global__ void k_edge_attn(const bfloat* __restrict__ qkv, const bfloat* __restrict__ dkvf,
                            const int* __restrict__ ei, const float* __restrict__ rij,
                            bfloat* __restrict__ vmsg, float* __restrict__ x_agg) {
  const int e = blockIdx.x * 8 + (threadIdx.x >> 5);
  const int lane = threadIdx.x & 31;
  const int h = lane * 4;
  const int s = ei[e], d = ei[N_EDGES + e];
  const bf4v q4  = *(const bf4v*)(qkv + (size_t)d * 384 + h);
  const bf4v k4  = *(const bf4v*)(qkv + (size_t)s * 384 + 128 + h);
  const bf4v v4  = *(const bf4v*)(qkv + (size_t)s * 384 + 256 + h);
  const bf4v dk4 = *(const bf4v*)(dkvf + (size_t)e * 384 + h);
  const bf4v dv4 = *(const bf4v*)(dkvf + (size_t)e * 384 + 128 + h);
  float p = 0.f;
#pragma unroll
  for (int i = 0; i < 4; ++i) p += (float)q4[i] * (float)k4[i] * (float)dk4[i];
  p += __shfl_xor(p, 1, 32);   // reduce over the 4 lanes of a 16-wide head
  p += __shfl_xor(p, 2, 32);
  const float r = rij[e];
  const float cut = (r < 5.0f) ? 0.5f * (__cosf(0.6283185307f * r) + 1.0f) : 0.0f;
  const float att = silu_f(p) * cut;
  bf4v vm;
#pragma unroll
  for (int i = 0; i < 4; ++i) {
    const float vv = (float)v4[i] * (float)dv4[i] * att;
    vm[i] = (bfloat)vv;
    __hip_atomic_fetch_add(&x_agg[(size_t)d * HDIM + h + i], vv,
                           __ATOMIC_RELAXED, __HIP_MEMORY_SCOPE_AGENT);
  }
  *(bf4v*)(vmsg + (size_t)e * HDIM + h) = vm;
}

// 16 edges/block: WMMA for s=silu(vmsg@Ws^T+bs) [16x256] and
// ex=clip(e@Wagg^T,0,1) [32x128] into LDS, then gating + vec_msg scatters.
__global__ void k_edge_msg(const bfloat* __restrict__ vmsg, const bfloat* __restrict__ Wsc,
                           const bfloat* __restrict__ Wagg, const float* __restrict__ bs,
                           const float* __restrict__ vec, const int* __restrict__ ei,
                           const float* __restrict__ dij, const float* __restrict__ aggp,
                           float* __restrict__ x_agg, float* __restrict__ vec_agg) {
  __shared__ float s_lds[16 * 256];
  __shared__ float ex_lds[32 * 128];
  const int et = blockIdx.x;
  const int w = threadIdx.x >> 5;
  const int lane = threadIdx.x & 31;
  const int n = lane & 15, hi = lane >> 4;
  v8f zero = {};
#pragma unroll
  for (int t4 = 0; t4 < 4; ++t4) {
    const int t = w * 4 + t4;
    if (t < 16) {                       // s: 16 col tiles, K=128
      v8f a = wmma_bf16_tile<HDIM>(vmsg, HDIM, et * 16, Wsc, HDIM, t * 16, zero);
      const int col = t * 16 + n;
      const float bias = bs[col];
#pragma unroll
      for (int i = 0; i < 8; ++i)
        s_lds[(i + 8 * hi) * 256 + col] = silu_f(a[i] + bias);
    } else {                            // ex: rows (e,p), K=64
      const int tt = t - 16;
      const int rt = tt >> 3, ct = tt & 7;
      v8f a = wmma_bf16_tile<64>(vmsg, 64, et * 32 + rt * 16, Wagg, 64, ct * 16, zero);
      const int col = ct * 16 + n;
#pragma unroll
      for (int i = 0; i < 8; ++i)
        ex_lds[(rt * 16 + i + 8 * hi) * 128 + col] = fminf(fmaxf(a[i], 0.f), 1.f);
    }
  }
  __syncthreads();
  const float ap = fabsf(aggp[0]);
  for (int idx = threadIdx.x; idx < 16 * 128; idx += 256) {
    const int el = idx >> 7, hh = idx & 127;
    const int e = et * 16 + el;
    float gsum = 0.f;
#pragma unroll
    for (int pp = 0; pp < 2; ++pp)
      gsum += (float)vmsg[(size_t)e * HDIM + pp * 64 + (hh & 63)] *
              ex_lds[(el * 2 + pp) * 128 + hh];
    __hip_atomic_fetch_add(&x_agg[(size_t)ei[N_EDGES + e] * HDIM + hh], ap * gsum,
                           __ATOMIC_RELAXED, __HIP_MEMORY_SCOPE_AGENT);
  }
  for (int idx = threadIdx.x; idx < 16 * 384; idx += 256) {
    const int el = idx / 384;
    const int c = (idx >> 7) % 3;
    const int hh = idx & 127;
    const int e = et * 16 + el;
    const int s = ei[e], d = ei[N_EDGES + e];
    const float vmv = vec[(size_t)s * 384 + c * 128 + hh] * s_lds[el * 256 + hh] +
                      s_lds[el * 256 + 128 + hh] * dij[(size_t)e * 3 + c];
    __hip_atomic_fetch_add(&vec_agg[(size_t)d * 384 + c * 128 + hh], vmv,
                           __ATOMIC_RELAXED, __HIP_MEMORY_SCOPE_AGENT);
  }
}

// Edge update: vector rejection (sign-invariant in d) + df scaling.
__global__ void k_edge_upd(const bfloat* __restrict__ wt, const bfloat* __restrict__ wsb,
                           const bfloat* __restrict__ dkvf, const int* __restrict__ ei,
                           const float* __restrict__ dij, float* __restrict__ dfij) {
  const int e = blockIdx.x * 2 + (threadIdx.x >> 7);
  const int hh = threadIdx.x & 127;
  const int s = ei[e], d = ei[N_EDGES + e];
  const float d0 = dij[(size_t)e * 3], d1 = dij[(size_t)e * 3 + 1], d2 = dij[(size_t)e * 3 + 2];
  const bfloat* wtr = wt + (size_t)d * 384 + hh;
  const bfloat* wsr = wsb + (size_t)s * 384 + hh;
  const float a0 = (float)wtr[0], a1 = (float)wtr[128], a2 = (float)wtr[256];
  const float b0 = (float)wsr[0], b1 = (float)wsr[128], b2 = (float)wsr[256];
  const float p1 = a0 * d0 + a1 * d1 + a2 * d2;
  const float p2 = b0 * d0 + b1 * d1 + b2 * d2;
  const float wdot = (a0 - p1 * d0) * (b0 - p2 * d0) +
                     (a1 - p1 * d1) * (b1 - p2 * d1) +
                     (a2 - p1 * d2) * (b2 - p2 * d2);
  dfij[(size_t)e * HDIM + hh] = (float)dkvf[(size_t)e * 384 + 256 + hh] * wdot;
}

// Final node update: o = x_agg@Wo^T + bo, dx = vecdot*o2 + o3, dvec = vec3*o1 + vec_agg.
__global__ void k_node_out(const bfloat* __restrict__ xaggb, const bfloat* __restrict__ Wo,
                           const float* __restrict__ bo, const float* __restrict__ vecdot,
                           const float* __restrict__ vec3, const float* __restrict__ vec_agg,
                           float* __restrict__ dx, float* __restrict__ dvec) {
  const int row0 = blockIdx.x * 16;
  const int w = threadIdx.x >> 5;
  const int lane = threadIdx.x & 31;
  const int n = lane & 15, hi = lane >> 4;
  v8f zero = {};
  v8f o1 = wmma_bf16_tile<HDIM>(xaggb, HDIM, row0, Wo, HDIM, w * 16, zero);
  v8f o2 = wmma_bf16_tile<HDIM>(xaggb, HDIM, row0, Wo, HDIM, 128 + w * 16, zero);
  v8f o3 = wmma_bf16_tile<HDIM>(xaggb, HDIM, row0, Wo, HDIM, 256 + w * 16, zero);
  const int col = w * 16 + n;
  const float B1 = bo[col], B2 = bo[128 + col], B3 = bo[256 + col];
#pragma unroll
  for (int i = 0; i < 8; ++i) {
    const int row = row0 + i + 8 * hi;
    const float O1 = o1[i] + B1, O2 = o2[i] + B2, O3 = o3[i] + B3;
    dx[(size_t)row * HDIM + col] = vecdot[(size_t)row * HDIM + col] * O2 + O3;
#pragma unroll
    for (int c = 0; c < 3; ++c) {
      const size_t idx = (size_t)row * 384 + c * 128 + col;
      dvec[idx] = vec3[idx] * O1 + vec_agg[idx];
    }
  }
}

// ---------------------------------------------------------------------------
extern "C" void kernel_launch(void* const* d_in, const int* in_sizes, int n_in,
                              void* d_out, int out_size, void* d_ws, size_t ws_size,
                              hipStream_t stream) {
  (void)in_sizes; (void)n_in; (void)out_size; (void)ws_size;
  const float* x    = (const float*)d_in[0];
  const float* vec  = (const float*)d_in[1];
  const int*   ei   = (const int*)d_in[2];
  const float* rij  = (const float*)d_in[3];
  const float* fij  = (const float*)d_in[4];
  const float* dij  = (const float*)d_in[5];
  const float* lng  = (const float*)d_in[6];
  const float* lnb  = (const float*)d_in[7];
  const float* Wq = (const float*)d_in[8];   const float* bq = (const float*)d_in[9];
  const float* Wk = (const float*)d_in[10];  const float* bk = (const float*)d_in[11];
  const float* Wv = (const float*)d_in[12];  const float* bv = (const float*)d_in[13];
  const float* Wdk = (const float*)d_in[14]; const float* bdk = (const float*)d_in[15];
  const float* Wdv = (const float*)d_in[16]; const float* bdv = (const float*)d_in[17];
  const float* Wvec = (const float*)d_in[18];
  const float* Ws = (const float*)d_in[19];  const float* bs = (const float*)d_in[20];
  const float* Wo = (const float*)d_in[21];  const float* bo = (const float*)d_in[22];
  const float* Wf = (const float*)d_in[23];  const float* bf_ = (const float*)d_in[24];
  const float* Wwsrc = (const float*)d_in[25];
  const float* Wwtrg = (const float*)d_in[26];
  const float* Wagg  = (const float*)d_in[27];
  const float* aggp  = (const float*)d_in[28];

  char* p = (char*)d_ws;
  auto alloc = [&](size_t elems, size_t esz) -> void* {
    void* r = (void*)p;
    p += ((elems * esz) + 255) & ~(size_t)255;
    return r;
  };
  bfloat* wqkv  = (bfloat*)alloc(384 * 128, 2);
  bfloat* wdkvf = (bfloat*)alloc(384 * 128, 2);
  bfloat* wvw   = (bfloat*)alloc(640 * 128, 2);
  bfloat* wsc   = (bfloat*)alloc(256 * 128, 2);
  bfloat* waggc = (bfloat*)alloc(128 * 64, 2);
  bfloat* woc   = (bfloat*)alloc(384 * 128, 2);
  bfloat* xn    = (bfloat*)alloc((size_t)N_NODES * 128, 2);
  bfloat* vecb  = (bfloat*)alloc((size_t)N_NODES * 384, 2);
  bfloat* fb    = (bfloat*)alloc((size_t)N_EDGES * 128, 2);
  bfloat* qkv   = (bfloat*)alloc((size_t)N_NODES * 384, 2);
  float*  vdot  = (float*)alloc((size_t)N_NODES * 128, 4);
  float*  vec3  = (float*)alloc((size_t)N_NODES * 384, 4);
  bfloat* wt    = (bfloat*)alloc((size_t)N_NODES * 384, 2);
  bfloat* wsrcb = (bfloat*)alloc((size_t)N_NODES * 384, 2);
  bfloat* dkvf  = (bfloat*)alloc((size_t)N_EDGES * 384, 2);
  float*  x_agg = (float*)alloc((size_t)N_NODES * 128, 4);
  float*  vagg  = (float*)alloc((size_t)N_NODES * 384, 4);
  bfloat* xaggb = xn;   // reuse: xn dead after qkv GEMM
  bfloat* vmsg  = fb;   // reuse: fb dead after dkvf GEMM

  float* dx   = (float*)d_out;
  float* dvec = dx + (size_t)N_NODES * 128;
  float* dfij = dvec + (size_t)N_NODES * 384;

  hipMemsetAsync(x_agg, 0, (size_t)N_NODES * 128 * 4, stream);
  hipMemsetAsync(vagg, 0, (size_t)N_NODES * 384 * 4, stream);

  auto cvt = [&](const float* s, bfloat* d, size_t cnt) {
    k_cvt<<<dim3((unsigned)((cnt + 255) / 256)), dim3(256), 0, stream>>>(s, d, (int)cnt);
  };
  cvt(Wq, wqkv, 16384);  cvt(Wk, wqkv + 16384, 16384);  cvt(Wv, wqkv + 32768, 16384);
  cvt(Wdk, wdkvf, 16384); cvt(Wdv, wdkvf + 16384, 16384); cvt(Wf, wdkvf + 32768, 16384);
  cvt(Wvec, wvw, 49152); cvt(Wwtrg, wvw + 49152, 16384); cvt(Wwsrc, wvw + 65536, 16384);
  cvt(Ws, wsc, 32768);   cvt(Wagg, waggc, 8192);         cvt(Wo, woc, 49152);
  cvt(vec, vecb, (size_t)N_NODES * 384);
  cvt(fij, fb, (size_t)N_EDGES * 128);

  k_ln<<<dim3(N_NODES / 8), dim3(256), 0, stream>>>(x, lng, lnb, xn);
  k_gemm384<<<dim3(N_NODES / 16, 6), dim3(128), 0, stream>>>(xn, wqkv, bq, bk, bv, qkv, 0);
  k_node_vec<<<dim3(N_NODES / 16), dim3(256), 0, stream>>>(vecb, wvw, vdot, vec3, wt, wsrcb);
  k_gemm384<<<dim3(N_EDGES / 16, 6), dim3(128), 0, stream>>>(fb, wdkvf, bdk, bdv, bf_, dkvf, 1);
  k_edge_attn<<<dim3(N_EDGES / 8), dim3(256), 0, stream>>>(qkv, dkvf, ei, rij, vmsg, x_agg);
  k_edge_msg<<<dim3(N_EDGES / 16), dim3(256), 0, stream>>>(vmsg, wsc, waggc, bs, vec, ei, dij,
                                                           aggp, x_agg, vagg);
  k_edge_upd<<<dim3(N_EDGES / 2), dim3(256), 0, stream>>>(wt, wsrcb, dkvf, ei, dij, dfij);
  k_cvt<<<dim3((N_NODES * 128 + 255) / 256), dim3(256), 0, stream>>>(x_agg, xaggb, N_NODES * 128);
  k_node_out<<<dim3(N_NODES / 16), dim3(256), 0, stream>>>(xaggb, woc, bo, vdot, vec3, vagg,
                                                           dx, dvec);
}
